// ModelB_Experts_25769804272
// MI455X (gfx1250) — compile-verified, matrix operand order checked by reading, not run on hardware
//
#include <hip/hip_runtime.h>
#include <hip/hip_bf16.h>
#include <math.h>

// ---------------- problem constants ----------------
#define B_    64
#define S_    512
#define N_    128
#define D_    512
#define H_    8
#define L_    4
#define FF_   2048
#define DH_   64
#define HID_  256
#define M_ROWS (B_ * N_)   // 8192

typedef __bf16 bf16_t;
typedef __attribute__((ext_vector_type(16))) __bf16 bf16x16;
typedef __attribute__((ext_vector_type(8)))  float  v8f;

// ---------------- small helpers ----------------
__device__ __forceinline__ float gelu_erf(float x) {
    return 0.5f * x * (1.0f + erff(x * 0.70710678118654752f));
}

// ---------------- f32 -> bf16 convert ----------------
__global__ __launch_bounds__(256) void cvt_bf16_kernel(const float* __restrict__ src,
                                                       bf16_t* __restrict__ dst, long n) {
    long i = (long)blockIdx.x * blockDim.x + threadIdx.x;
    long stride = (long)gridDim.x * blockDim.x;
    for (; i < n; i += stride) dst[i] = (bf16_t)src[i];
}

// exp1_w [N, D, HID] -> [N, HID, D] bf16 (weight-major for WMMA B operand)
__global__ __launch_bounds__(256) void cvt_exp1_kernel(const float* __restrict__ src,
                                                       bf16_t* __restrict__ dst) {
    long n_tot = (long)N_ * D_ * HID_;
    long i = (long)blockIdx.x * blockDim.x + threadIdx.x;
    long stride = (long)gridDim.x * blockDim.x;
    for (; i < n_tot; i += stride) {
        long n = i / ((long)D_ * HID_);
        long rem = i - n * (long)D_ * HID_;
        long d = rem / HID_;
        long h = rem - d * HID_;
        dst[n * (long)HID_ * D_ + h * D_ + d] = (bf16_t)src[i];
    }
}

// ---------------- encoder ----------------
__global__ __launch_bounds__(256) void encoder_kernel(
    const float* __restrict__ base, const float* __restrict__ intr,
    const float* __restrict__ target, const float* __restrict__ mask,
    const float* __restrict__ enc_w, const float* __restrict__ enc_b,
    const float* __restrict__ node_emb,
    float* __restrict__ xf, bf16_t* __restrict__ xb)
{
    int bn = blockIdx.x;
    int b = bn / N_, n = bn % N_;
    int tid = threadIdx.x;

    float s1 = 0.f, q1 = 0.f, s2 = 0.f, q2 = 0.f;
    for (int s = tid; s < S_; s += 256) {
        float a = base[((long)b * S_ + s) * N_ + n];
        float c = intr[((long)b * S_ + s) * N_ + n];
        s1 += a; q1 += a * a; s2 += c; q2 += c * c;
    }
    __shared__ float red[4][8];
    int wave = tid >> 5, lane = tid & 31;
    float vals[4] = {s1, q1, s2, q2};
#pragma unroll
    for (int i = 0; i < 4; ++i) {
        float v = vals[i];
        for (int m = 1; m < 32; m <<= 1) v += __shfl_xor(v, m, 32);
        if (lane == 0) red[i][wave] = v;
    }
    __syncthreads();
    __shared__ float feat[6];
    if (tid == 0) {
        float t1 = 0.f, t2 = 0.f, t3 = 0.f, t4 = 0.f;
#pragma unroll
        for (int i = 0; i < 8; ++i) { t1 += red[0][i]; t2 += red[1][i]; t3 += red[2][i]; t4 += red[3][i]; }
        float inv = 1.0f / (float)S_;
        float m1 = t1 * inv, m2 = t3 * inv;
        feat[0] = m1;
        feat[1] = sqrtf(fmaxf(t2 * inv - m1 * m1, 0.f));
        feat[2] = m2;
        feat[3] = sqrtf(fmaxf(t4 * inv - m2 * m2, 0.f));
        feat[4] = target[(long)b * N_ + n];
        feat[5] = mask[(long)b * N_ + n];
    }
    __syncthreads();
    for (int d = tid; d < D_; d += 256) {
        float acc = enc_b[d] + node_emb[(long)n * D_ + d];
#pragma unroll
        for (int f = 0; f < 6; ++f) acc += feat[f] * enc_w[f * D_ + d];
        xf[(long)bn * D_ + d] = acc;
        xb[(long)bn * D_ + d] = (bf16_t)acc;
    }
}

// ---------------- generic batched WMMA GEMM (double-buffered LDS) ----------------
// C[z][m][n] = sum_k A[z][m][k] * W[z][n][k]  (+ bias[z][n]); optional GELU;
// writes f32 (Cf) and/or bf16 (Cb). Wave tile = 16 rows x 128 cols (8 WMMA/K-step).
#define BM 128
#define BN 128
#define BK 32

__global__ __launch_bounds__(256) void gemm_bf16_kernel(
    const bf16_t* __restrict__ A, long lda, long Abatch,
    const bf16_t* __restrict__ W, long Wbatch,
    const float* __restrict__ bias, long biasBatch,
    float* __restrict__ Cf, bf16_t* __restrict__ Cb, long ldc, long Cbatch,
    int M, int N, int K, int gelu_flag)
{
    __shared__ __align__(64) bf16_t sA[2][BM * BK];   // 2 x 8 KB
    __shared__ __align__(64) bf16_t sB[2][BN * BK];   // 2 x 8 KB

    const int tid  = threadIdx.x;
    const int wave = tid >> 5;
    const int lane = tid & 31;
    const int m0 = blockIdx.y * BM;
    const int n0 = blockIdx.x * BN;
    const long zb = blockIdx.z;

    const bf16_t* Ab = A + zb * Abatch;
    const bf16_t* Wb = W + zb * Wbatch;

    // staging assignment: thread covers rows r0 and r0+64 of both tiles, 8 bf16 each
    const int r0 = tid >> 2;
    const int o0 = (tid & 3) * 8;
    const bool fullM = (m0 + BM <= M);
    const bool okA0 = fullM || (m0 + r0 < M);
    const bool okA1 = fullM || (m0 + r0 + 64 < M);

    v8f acc[8];
#pragma unroll
    for (int t = 0; t < 8; ++t) acc[t] = (v8f){0.f,0.f,0.f,0.f,0.f,0.f,0.f,0.f};

    const uint4 zero4 = make_uint4(0u, 0u, 0u, 0u);
    uint4 pa0, pa1, pb0, pb1;

    // prologue: prefetch K-tile 0
    pa0 = okA0 ? *(const uint4*)(Ab + (long)(m0 + r0) * lda + o0) : zero4;
    pa1 = okA1 ? *(const uint4*)(Ab + (long)(m0 + r0 + 64) * lda + o0) : zero4;
    pb0 = *(const uint4*)(Wb + (long)(n0 + r0) * K + o0);
    pb1 = *(const uint4*)(Wb + (long)(n0 + r0 + 64) * K + o0);

    const int nsteps = K / BK;
    for (int s = 0; s < nsteps; ++s) {
        const int buf = s & 1;
        // commit prefetched tile to LDS
        *(uint4*)(&sA[buf][r0 * BK + o0])        = pa0;
        *(uint4*)(&sA[buf][(r0 + 64) * BK + o0]) = pa1;
        *(uint4*)(&sB[buf][r0 * BK + o0])        = pb0;
        *(uint4*)(&sB[buf][(r0 + 64) * BK + o0]) = pb1;
        __syncthreads();

        // prefetch next K-tile from global while WMMAs run
        if (s + 1 < nsteps) {
            const int k0 = (s + 1) * BK;
            pa0 = okA0 ? *(const uint4*)(Ab + (long)(m0 + r0) * lda + k0 + o0) : zero4;
            pa1 = okA1 ? *(const uint4*)(Ab + (long)(m0 + r0 + 64) * lda + k0 + o0) : zero4;
            pb0 = *(const uint4*)(Wb + (long)(n0 + r0) * K + k0 + o0);
            pb1 = *(const uint4*)(Wb + (long)(n0 + r0 + 64) * K + k0 + o0);
        }

        bf16x16 afrag = *(const bf16x16*)(&sA[buf][(wave * 16 + (lane & 15)) * BK + (lane >> 4) * 16]);
#pragma unroll
        for (int t = 0; t < 8; ++t) {
            bf16x16 bfrag = *(const bf16x16*)(&sB[buf][(t * 16 + (lane & 15)) * BK + (lane >> 4) * 16]);
            acc[t] = __builtin_amdgcn_wmma_f32_16x16x32_bf16(
                false, afrag, false, bfrag, (short)0, acc[t], false, false);
        }
        // no trailing barrier: next store goes to the other buffer; the barrier at
        // the top of iteration s+1 orders reads(buf) vs writes(buf) two steps later
    }

    // epilogue
#pragma unroll
    for (int t = 0; t < 8; ++t) {
        int n = n0 + t * 16 + (lane & 15);
        float bv = bias ? bias[zb * biasBatch + n] : 0.f;
#pragma unroll
        for (int j = 0; j < 8; ++j) {
            int m = m0 + wave * 16 + j + ((lane >> 4) << 3);
            if (fullM || m < M) {
                float v = acc[t][j] + bv;
                if (gelu_flag) v = gelu_erf(v);
                long idx = zb * Cbatch + (long)m * ldc + n;
                if (Cf) Cf[idx] = v;
                if (Cb) Cb[idx] = (bf16_t)v;
            }
        }
    }
}

// ---------------- attention: one workgroup per (b,h) ----------------
__global__ __launch_bounds__(256) void attn_kernel(const float* __restrict__ qkv,
                                                   bf16_t* __restrict__ o)
{
    __shared__ __align__(64) bf16_t sQK[2 * N_ * DH_]; // Q|K, later reused as P (128x128)
    __shared__ __align__(64) bf16_t sVT[DH_ * N_];     // V transposed [d][n]

    int bh = blockIdx.x;
    int b = bh / H_, h = bh % H_;
    const float* basep = qkv + (long)b * N_ * (3 * D_);
    int tid = threadIdx.x;
    int wave = tid >> 5, lane = tid & 31;

    for (int i = tid; i < N_ * DH_; i += 256) {
        int n = i >> 6, d = i & 63;
        long row = (long)n * (3 * D_) + h * DH_ + d;
        sQK[i]                 = (bf16_t)basep[row];
        sQK[N_ * DH_ + i]      = (bf16_t)basep[row + D_];
        sVT[d * N_ + n]        = (bf16_t)basep[row + 2 * D_];
    }
    __syncthreads();

    // S = Q @ K^T  (wave strip: rows wave*16..+15, all 128 cols -> 8 tiles)
    v8f acc[8];
#pragma unroll
    for (int t = 0; t < 8; ++t) acc[t] = (v8f){0.f,0.f,0.f,0.f,0.f,0.f,0.f,0.f};
    for (int k0 = 0; k0 < DH_; k0 += BK) {
        bf16x16 afrag = *(const bf16x16*)(&sQK[(wave * 16 + (lane & 15)) * DH_ + k0 + (lane >> 4) * 16]);
#pragma unroll
        for (int t = 0; t < 8; ++t) {
            bf16x16 bfrag = *(const bf16x16*)(&sQK[N_ * DH_ + (t * 16 + (lane & 15)) * DH_ + k0 + (lane >> 4) * 16]);
            acc[t] = __builtin_amdgcn_wmma_f32_16x16x32_bf16(
                false, afrag, false, bfrag, (short)0, acc[t], false, false);
        }
    }

    // row softmax in registers; C layout: elem(m,n): vgpr j, m = wave*16+j+8*(lane>>4), n = t*16+(lane&15)
    const float scale = 0.125f; // 1/sqrt(DH)
    float p[8][8];
#pragma unroll
    for (int j = 0; j < 8; ++j) {
        float mx = -3.0e38f;
#pragma unroll
        for (int t = 0; t < 8; ++t) mx = fmaxf(mx, acc[t][j]);
        for (int m = 1; m <= 8; m <<= 1) mx = fmaxf(mx, __shfl_xor(mx, m, 32));
        float sum = 0.f;
#pragma unroll
        for (int t = 0; t < 8; ++t) {
            float e = __expf((acc[t][j] - mx) * scale);
            p[t][j] = e; sum += e;
        }
        for (int m = 1; m <= 8; m <<= 1) sum += __shfl_xor(sum, m, 32);
        float inv = 1.0f / sum;
#pragma unroll
        for (int t = 0; t < 8; ++t) p[t][j] *= inv;
    }
    __syncthreads(); // everyone done reading Q/K from LDS

    // write P (bf16) into sQK reused as [128][128]
#pragma unroll
    for (int t = 0; t < 8; ++t) {
        int n = t * 16 + (lane & 15);
#pragma unroll
        for (int j = 0; j < 8; ++j) {
            int m = wave * 16 + j + ((lane >> 4) << 3);
            sQK[m * N_ + n] = (bf16_t)p[t][j];
        }
    }
    __syncthreads();

    // O = P @ V  (K-dim = 128 nodes, N-dim = 64 head dims)
    v8f oacc[4];
#pragma unroll
    for (int t = 0; t < 4; ++t) oacc[t] = (v8f){0.f,0.f,0.f,0.f,0.f,0.f,0.f,0.f};
    for (int k0 = 0; k0 < N_; k0 += BK) {
        bf16x16 afrag = *(const bf16x16*)(&sQK[(wave * 16 + (lane & 15)) * N_ + k0 + (lane >> 4) * 16]);
#pragma unroll
        for (int t = 0; t < 4; ++t) {
            bf16x16 bfrag = *(const bf16x16*)(&sVT[(t * 16 + (lane & 15)) * N_ + k0 + (lane >> 4) * 16]);
            oacc[t] = __builtin_amdgcn_wmma_f32_16x16x32_bf16(
                false, afrag, false, bfrag, (short)0, oacc[t], false, false);
        }
    }
#pragma unroll
    for (int t = 0; t < 4; ++t) {
        int d = t * 16 + (lane & 15);
#pragma unroll
        for (int j = 0; j < 8; ++j) {
            int m = wave * 16 + j + ((lane >> 4) << 3);
            o[((long)(b * N_ + m)) * D_ + h * DH_ + d] = (bf16_t)oacc[t][j];
        }
    }
}

// ---------------- residual add + layernorm ----------------
__global__ __launch_bounds__(256) void add_ln_kernel(
    const float* __restrict__ x, const float* __restrict__ dlt,
    const float* __restrict__ sc, const float* __restrict__ bi,
    float* __restrict__ xf, bf16_t* __restrict__ xb)
{
    int row = blockIdx.x;
    int tid = threadIdx.x;
    int wave = tid >> 5, lane = tid & 31;
    __shared__ float redA[8], redB[8];

    float v[2];
    float sum = 0.f, sq = 0.f;
#pragma unroll
    for (int i = 0; i < 2; ++i) {
        int d = tid + i * 256;
        float t = x[(long)row * D_ + d] + dlt[(long)row * D_ + d];
        v[i] = t; sum += t; sq += t * t;
    }
    for (int m = 1; m < 32; m <<= 1) { sum += __shfl_xor(sum, m, 32); sq += __shfl_xor(sq, m, 32); }
    if (lane == 0) { redA[wave] = sum; redB[wave] = sq; }
    __syncthreads();
    float tsum = 0.f, tsq = 0.f;
#pragma unroll
    for (int i = 0; i < 8; ++i) { tsum += redA[i]; tsq += redB[i]; }
    float mu = tsum * (1.0f / D_);
    float var = tsq * (1.0f / D_) - mu * mu;
    float rstd = rsqrtf(var + 1e-5f);
#pragma unroll
    for (int i = 0; i < 2; ++i) {
        int d = tid + i * 256;
        float y = (v[i] - mu) * rstd * sc[d] + bi[d];
        xf[(long)row * D_ + d] = y;
        xb[(long)row * D_ + d] = (bf16_t)y;
    }
}

// ---------------- expert deltas: dot(eh[b,n,:], exp2_w[n,:]) + b ----------------
__global__ __launch_bounds__(256) void deltas_kernel(
    const bf16_t* __restrict__ eh, const float* __restrict__ w2,
    const float* __restrict__ b2, float* __restrict__ out)
{
    int idx = blockIdx.x * 8 + (threadIdx.x >> 5);   // row = b*N + n
    int lane = threadIdx.x & 31;
    int n = idx % N_;
    const bf16_t* e = eh + (long)idx * HID_;
    const float*  w = w2 + (long)n * HID_;
    float sum = 0.f;
    for (int i = lane; i < HID_; i += 32) sum += (float)e[i] * w[i];
    for (int m = 1; m < 32; m <<= 1) sum += __shfl_xor(sum, m, 32);
    if (lane == 0) out[idx] = sum + b2[n];
}

// ---------------- host orchestration ----------------
extern "C" void kernel_launch(void* const* d_in, const int* in_sizes, int n_in,
                              void* d_out, int out_size, void* d_ws, size_t ws_size,
                              hipStream_t stream)
{
    (void)in_sizes; (void)n_in; (void)out_size; (void)ws_size;
    const float* base_samples = (const float*)d_in[0];
    const float* int_samples  = (const float*)d_in[1];
    const float* target_row   = (const float*)d_in[2];
    const float* int_mask     = (const float*)d_in[3];
    const float* enc_w        = (const float*)d_in[4];
    const float* enc_b        = (const float*)d_in[5];
    const float* node_emb     = (const float*)d_in[6];
    const float* qkv_w        = (const float*)d_in[7];
    const float* qkv_b        = (const float*)d_in[8];
    const float* out_w        = (const float*)d_in[9];
    const float* out_b        = (const float*)d_in[10];
    const float* ln1_s        = (const float*)d_in[11];
    const float* ln1_b        = (const float*)d_in[12];
    const float* ff1_w        = (const float*)d_in[13];
    const float* ff1_b        = (const float*)d_in[14];
    const float* ff2_w        = (const float*)d_in[15];
    const float* ff2_b        = (const float*)d_in[16];
    const float* ln2_s        = (const float*)d_in[17];
    const float* ln2_b        = (const float*)d_in[18];
    const float* exp1_w       = (const float*)d_in[19];
    const float* exp1_b       = (const float*)d_in[20];
    const float* exp2_w       = (const float*)d_in[21];
    const float* exp2_b       = (const float*)d_in[22];
    const float* dagp_w       = (const float*)d_in[23];
    const float* dagp_b       = (const float*)d_in[24];
    const float* dagc_w       = (const float*)d_in[25];
    const float* dagc_b       = (const float*)d_in[26];

    float* out_f = (float*)d_out;      // [0,8192): deltas ; [8192, +B*N*N): adj

    // ---- workspace carve ----
    char* wp = (char*)d_ws;
    auto carve = [&](size_t bytes) -> char* {
        char* p = wp; wp += (bytes + 255) & ~(size_t)255; return p;
    };
    bf16_t* wb_qkv  = (bf16_t*)carve((size_t)L_ * 3 * D_ * D_ * 2);
    bf16_t* wb_out  = (bf16_t*)carve((size_t)L_ * D_ * D_ * 2);
    bf16_t* wb_ff1  = (bf16_t*)carve((size_t)L_ * FF_ * D_ * 2);
    bf16_t* wb_ff2  = (bf16_t*)carve((size_t)L_ * D_ * FF_ * 2);
    bf16_t* wb_exp1 = (bf16_t*)carve((size_t)N_ * HID_ * D_ * 2);
    bf16_t* wb_dagp = (bf16_t*)carve((size_t)D_ * D_ * 2);
    bf16_t* wb_dagc = (bf16_t*)carve((size_t)D_ * D_ * 2);
    float*  xf      = (float*) carve((size_t)M_ROWS * D_ * 4);
    bf16_t* xb      = (bf16_t*)carve((size_t)M_ROWS * D_ * 2);
    float*  qkvf    = (float*) carve((size_t)M_ROWS * 3 * D_ * 4);
    bf16_t* ob      = (bf16_t*)carve((size_t)M_ROWS * D_ * 2);
    float*  tmpf    = (float*) carve((size_t)M_ROWS * D_ * 4);
    bf16_t* h1b     = (bf16_t*)carve((size_t)M_ROWS * FF_ * 2);
    bf16_t* ehb     = (bf16_t*)carve((size_t)M_ROWS * HID_ * 2);
    bf16_t* pb      = (bf16_t*)carve((size_t)M_ROWS * D_ * 2);
    bf16_t* cb      = (bf16_t*)carve((size_t)M_ROWS * D_ * 2);

    // ---- weight conversion ----
    cvt_bf16_kernel<<<2048, 256, 0, stream>>>(qkv_w, wb_qkv, (long)L_ * 3 * D_ * D_);
    cvt_bf16_kernel<<<2048, 256, 0, stream>>>(out_w, wb_out, (long)L_ * D_ * D_);
    cvt_bf16_kernel<<<2048, 256, 0, stream>>>(ff1_w, wb_ff1, (long)L_ * FF_ * D_);
    cvt_bf16_kernel<<<2048, 256, 0, stream>>>(ff2_w, wb_ff2, (long)L_ * D_ * FF_);
    cvt_bf16_kernel<<<512,  256, 0, stream>>>(dagp_w, wb_dagp, (long)D_ * D_);
    cvt_bf16_kernel<<<512,  256, 0, stream>>>(dagc_w, wb_dagc, (long)D_ * D_);
    cvt_exp1_kernel<<<4096, 256, 0, stream>>>(exp1_w, wb_exp1);

    // ---- encoder ----
    encoder_kernel<<<M_ROWS, 256, 0, stream>>>(base_samples, int_samples, target_row,
                                               int_mask, enc_w, enc_b, node_emb, xf, xb);

    // ---- transformer layers ----
    for (int l = 0; l < L_; ++l) {
        // QKV: [8192,512] @ [1536,512]^T -> qkvf
        gemm_bf16_kernel<<<dim3(3 * D_ / BN, M_ROWS / BM, 1), 256, 0, stream>>>(
            xb, D_, 0, wb_qkv + (long)l * 3 * D_ * D_, 0,
            qkv_b + (long)l * 3 * D_, 0,
            qkvf, nullptr, 3 * D_, 0, M_ROWS, 3 * D_, D_, 0);
        // attention per (b,h)
        attn_kernel<<<B_ * H_, 256, 0, stream>>>(qkvf, ob);
        // out-proj
        gemm_bf16_kernel<<<dim3(D_ / BN, M_ROWS / BM, 1), 256, 0, stream>>>(
            ob, D_, 0, wb_out + (long)l * D_ * D_, 0,
            out_b + (long)l * D_, 0,
            tmpf, nullptr, D_, 0, M_ROWS, D_, D_, 0);
        add_ln_kernel<<<M_ROWS, 256, 0, stream>>>(xf, tmpf, ln1_s + (long)l * D_,
                                                  ln1_b + (long)l * D_, xf, xb);
        // FF1 (+gelu, bf16 out)
        gemm_bf16_kernel<<<dim3(FF_ / BN, M_ROWS / BM, 1), 256, 0, stream>>>(
            xb, D_, 0, wb_ff1 + (long)l * FF_ * D_, 0,
            ff1_b + (long)l * FF_, 0,
            nullptr, h1b, FF_, 0, M_ROWS, FF_, D_, 1);
        // FF2
        gemm_bf16_kernel<<<dim3(D_ / BN, M_ROWS / BM, 1), 256, 0, stream>>>(
            h1b, FF_, 0, wb_ff2 + (long)l * D_ * FF_, 0,
            ff2_b + (long)l * D_, 0,
            tmpf, nullptr, D_, 0, M_ROWS, D_, FF_, 0);
        add_ln_kernel<<<M_ROWS, 256, 0, stream>>>(xf, tmpf, ln2_s + (long)l * D_,
                                                  ln2_b + (long)l * D_, xf, xb);
    }

    // ---- per-node experts: batched over n (z dim), M=B=64 rows ----
    gemm_bf16_kernel<<<dim3(HID_ / BN, 1, N_), 256, 0, stream>>>(
        xb, (long)N_ * D_, (long)D_,                 // A: row b -> xb[(b*N+n)*D]
        wb_exp1, (long)HID_ * D_,
        exp1_b, (long)HID_,
        nullptr, ehb, (long)N_ * HID_, (long)HID_,   // C: eh[(b*N+n)*HID + h]
        B_, HID_, D_, 1);
    deltas_kernel<<<M_ROWS / 8, 256, 0, stream>>>(ehb, exp2_w, exp2_b, out_f);

    // ---- DAG projections ----
    gemm_bf16_kernel<<<dim3(D_ / BN, M_ROWS / BM, 1), 256, 0, stream>>>(
        xb, D_, 0, wb_dagp, 0, dagp_b, 0,
        nullptr, pb, D_, 0, M_ROWS, D_, D_, 0);
    gemm_bf16_kernel<<<dim3(D_ / BN, M_ROWS / BM, 1), 256, 0, stream>>>(
        xb, D_, 0, wb_dagc, 0, dagc_b, 0,
        nullptr, cb, D_, 0, M_ROWS, D_, D_, 0);

    // ---- adjacency: per-batch p[b] (128x512) @ c[b]^T (512x128) ----
    gemm_bf16_kernel<<<dim3(N_ / BN, 1, B_), 256, 0, stream>>>(
        pb, D_, (long)N_ * D_,
        cb, (long)N_ * D_,
        nullptr, 0,
        out_f + M_ROWS, nullptr, N_, (long)N_ * N_,
        N_, N_, D_, 0);
}
